// ModalityAwareDualAttention_63050119905891
// MI455X (gfx1250) — matrix operand order repaired and checked by reading, not compiled, not run on hardware
//
#include <hip/hip_runtime.h>
#include <hip/hip_bf16.h>
#include <stdint.h>

typedef __attribute__((ext_vector_type(16))) _Float16 v16h;
typedef __attribute__((ext_vector_type(8)))  float    v8f;
typedef __attribute__((ext_vector_type(4)))  unsigned int u32x4;
typedef __attribute__((ext_vector_type(8)))  int      i32x8;
typedef __attribute__((ext_vector_type(4)))  int      i32x4;

#define DEVINL __device__ __forceinline__

#if defined(__has_builtin)
#if __has_builtin(__builtin_amdgcn_tensor_load_to_lds) && __has_builtin(__builtin_amdgcn_s_wait_tensorcnt)
#define HAVE_TDM 1
#endif
#endif

// ---- problem constants -------------------------------------------------
constexpr int B_  = 32;
constexpr int C_  = 1024;
constexpr int H_  = 48;
constexpr int W_  = 24;
constexpr int P_  = 3;
constexpr int IC_ = 128;          // C/8
constexpr int PH_ = 16;           // H/P
constexpr int N_  = 384;          // PH*W
constexpr int HW_ = H_ * W_;      // 1152
constexpr int NI_ = B_ * P_;      // 96 instances

DEVINL int laneid() { return (int)(threadIdx.x & 31u); }

DEVINL v8f wmma(v16h a, v16h b, v8f c) {
  return __builtin_amdgcn_wmma_f32_16x16x32_f16(false, a, false, b, (short)0, c,
                                                false, false);
}

// A fragment (16x32, M x K) from row-major f16, K contiguous with stride ldk.
// lane<16: K 0..15; lane>=16: K 16..31 (16 contiguous halves).
DEVINL v16h loadA_f16(const _Float16* m, int ldk, int row0, int k0) {
  int l = laneid();
  const _Float16* r = m + (size_t)(row0 + (l & 15)) * ldk + k0 + ((l < 16) ? 0 : 16);
  v16h f;
#pragma unroll
  for (int i = 0; i < 16; ++i) f[i] = r[i];
  return f;
}

// B fragment (32x16, K x N) given B^T row-major [col][k], K contiguous.
// lane<16: K 0..15; lane>=16: K 16..31 (16 contiguous halves).
DEVINL v16h loadB_f16(const _Float16* m, int ldk, int col0, int k0) {
  int l = laneid();
  const _Float16* r = m + (size_t)(col0 + (l & 15)) * ldk + k0 + ((l < 16) ? 0 : 16);
  v16h f;
#pragma unroll
  for (int i = 0; i < 16; ++i) f[i] = r[i];
  return f;
}

// B fragment from row-major f32 (K contiguous), converted on the fly.
DEVINL v16h loadB_f32(const float* m, size_t ldk, int col0, int k0) {
  int l = laneid();
  const float* r = m + (size_t)(col0 + (l & 15)) * ldk + k0 + ((l < 16) ? 0 : 16);
  v16h f;
#pragma unroll
  for (int i = 0; i < 16; ++i) f[i] = (_Float16)r[i];
  return f;
}

// ---- kernel 1: modality gate MLP --------------------------------------
__global__ void gate_kernel(const int* __restrict__ modality,
                            const float* __restrict__ gW1, const float* __restrict__ gb1,
                            const float* __restrict__ gW2, const float* __restrict__ gb2,
                            float* __restrict__ gate) {
  int t = (int)threadIdx.x;
  if (t >= NI_) return;
  int b = t / P_, p = t % P_;
  float m = (float)modality[b];
  float acc = gb2[p];
#pragma unroll 8
  for (int j = 0; j < 64; ++j) {
    float h = m * gW1[j] + gb1[j];
    h = h > 0.f ? h : 0.f;
    acc += gW2[p * 64 + j] * h;
  }
  gate[t] = 1.f / (1.f + __expf(-acc));
}

// ---- kernel 2: generic f32 -> f16 conversion (for Wq/Wk/Wv) -----------
__global__ void cvt_f32_f16_kernel(const float* __restrict__ src,
                                   _Float16* __restrict__ dst, int n) {
  int i = (int)(blockIdx.x * blockDim.x + threadIdx.x);
  int stride = (int)(gridDim.x * blockDim.x);
  for (; i < n; i += stride) dst[i] = (_Float16)src[i];
}

// ---- kernel 3: stage xf^T as f16 [inst][n][c] via LDS transpose -------
// grid(16, 6, 96), block 256. 64c x 64n tile per block.
__global__ void stage_xT_kernel(const float* __restrict__ x,
                                _Float16* __restrict__ xT) {
  __shared__ _Float16 T[64][72];
  int ctile = (int)blockIdx.x, ntile = (int)blockIdx.y, inst = (int)blockIdx.z;
  int b = inst / P_, p = inst % P_;
  const float* xpart = x + (size_t)b * C_ * HW_ + (size_t)p * PH_ * W_;
  int c0 = ctile * 64, n0 = ntile * 64;
  int tid = (int)threadIdx.x;
  for (int i = tid; i < 64 * 64; i += 256) {
    int r = i >> 6, n = i & 63;
    T[r][n] = (_Float16)xpart[(size_t)(c0 + r) * HW_ + n0 + n];
  }
  __syncthreads();
  _Float16* dst = xT + (size_t)inst * N_ * C_;
  for (int i = tid; i < 64 * 64; i += 256) {
    int nn = i >> 6, cc = i & 63;
    dst[(size_t)(n0 + nn) * C_ + c0 + cc] = T[cc][nn];
  }
}

// ---- kernel 4: q / kT projection  out[n][o] = sum_c W[o][c]*xfT[n][c]+b[o]
// grid(6, 96), block 128 (4 waves). All-f16 operands.
__global__ void proj_qk_kernel(const _Float16* __restrict__ xT,
                               const _Float16* __restrict__ W16,
                               const float* __restrict__ bias,
                               _Float16* __restrict__ out) {
  int mt = (int)blockIdx.x, inst = (int)blockIdx.y;
  int p = inst % P_;
  int wave = (int)(threadIdx.x >> 5);
  const _Float16* xTp = xT + (size_t)inst * N_ * C_;
  const _Float16* Wp  = W16 + (size_t)p * IC_ * C_;
  int m0 = mt * 64 + wave * 16;

  v8f acc[8];
#pragma unroll
  for (int t = 0; t < 8; ++t)
#pragma unroll
    for (int j = 0; j < 8; ++j) acc[t][j] = 0.f;

  for (int k0 = 0; k0 < C_; k0 += 32) {
    v16h a = loadA_f16(xTp, C_, m0, k0);
#pragma unroll
    for (int t = 0; t < 8; ++t) {
      v16h bb = loadB_f16(Wp, C_, t * 16, k0);
      acc[t] = wmma(a, bb, acc[t]);
    }
  }
  int l = laneid(), colb = l & 15, rofs = (l < 16) ? 0 : 8;
  _Float16* op = out + (size_t)inst * N_ * IC_;
#pragma unroll
  for (int t = 0; t < 8; ++t) {
    float bv = bias[p * IC_ + t * 16 + colb];
#pragma unroll
    for (int j = 0; j < 8; ++j) {
      int row = m0 + j + rofs;
      op[(size_t)row * IC_ + t * 16 + colb] = (_Float16)(acc[t][j] + bv);
    }
  }
}

// ---- kernel 5: attention S = q·kT, row softmax, store f16 -------------
// grid(12, 96), block 64 (2 waves). LDS 48KB.
__global__ void attn_kernel(const _Float16* __restrict__ q,
                            const _Float16* __restrict__ kT,
                            _Float16* __restrict__ attn) {
  __shared__ float S[2][16][N_];
  int mt = (int)blockIdx.x, inst = (int)blockIdx.y;
  int wave = (int)(threadIdx.x >> 5), l = laneid();
  int colb = l & 15, rofs = (l < 16) ? 0 : 8;
  const _Float16* qp = q + (size_t)inst * N_ * IC_;
  const _Float16* kp = kT + (size_t)inst * N_ * IC_;
  int m0 = mt * 32 + wave * 16;

  for (int nt = 0; nt < 24; ++nt) {
    v8f acc;
#pragma unroll
    for (int j = 0; j < 8; ++j) acc[j] = 0.f;
#pragma unroll
    for (int k0 = 0; k0 < IC_; k0 += 32) {
      v16h a  = loadA_f16(qp, IC_, m0, k0);
      v16h bb = loadB_f16(kp, IC_, nt * 16, k0);
      acc = wmma(a, bb, acc);
    }
#pragma unroll
    for (int j = 0; j < 8; ++j) S[wave][j + rofs][nt * 16 + colb] = acc[j];
  }
  _Float16* ap = attn + (size_t)inst * N_ * N_;
  for (int r = 0; r < 16; ++r) {
    float mx = -3.4e38f;
    for (int i = l; i < N_; i += 32) mx = fmaxf(mx, S[wave][r][i]);
#pragma unroll
    for (int d = 16; d; d >>= 1) mx = fmaxf(mx, __shfl_xor(mx, d, 32));
    float sm = 0.f;
    for (int i = l; i < N_; i += 32) sm += __expf(S[wave][r][i] - mx);
#pragma unroll
    for (int d = 16; d; d >>= 1) sm += __shfl_xor(sm, d, 32);
    float inv = 1.f / sm;
    for (int i = l; i < N_; i += 32)
      ap[(size_t)(m0 + r) * N_ + i] = (_Float16)(__expf(S[wave][r][i] - mx) * inv);
  }
}

// ---- kernel 6: v projection  v[c][n] = sum_cc Wv[c][cc]*xfT[n][cc]+bv[c]
// grid(16, 3, 96), block 128. All-f16 operands.
__global__ void proj_v_kernel(const _Float16* __restrict__ xT,
                              const _Float16* __restrict__ Wv16,
                              const float* __restrict__ bv,
                              _Float16* __restrict__ vout) {
  int mt = (int)blockIdx.x, nc = (int)blockIdx.y, inst = (int)blockIdx.z;
  int p = inst % P_;
  int wave = (int)(threadIdx.x >> 5);
  const _Float16* xTp = xT + (size_t)inst * N_ * C_;
  const _Float16* Wp  = Wv16 + (size_t)p * C_ * C_;
  int m0 = mt * 64 + wave * 16;   // c rows
  int n0 = nc * 128;

  v8f acc[8];
#pragma unroll
  for (int t = 0; t < 8; ++t)
#pragma unroll
    for (int j = 0; j < 8; ++j) acc[t][j] = 0.f;

  for (int k0 = 0; k0 < C_; k0 += 32) {
    __builtin_prefetch(Wp + (size_t)m0 * C_ + k0 + 128, 0, 0);
    v16h a = loadA_f16(Wp, C_, m0, k0);
#pragma unroll
    for (int t = 0; t < 8; ++t) {
      v16h bb = loadB_f16(xTp, C_, n0 + t * 16, k0);
      acc[t] = wmma(a, bb, acc[t]);
    }
  }
  int l = laneid(), colb = l & 15, rofs = (l < 16) ? 0 : 8;
  _Float16* vp = vout + (size_t)inst * C_ * N_;
#pragma unroll
  for (int t = 0; t < 8; ++t)
#pragma unroll
    for (int j = 0; j < 8; ++j) {
      int c = m0 + j + rofs, n = n0 + t * 16 + colb;
      vp[(size_t)c * N_ + n] = (_Float16)(acc[t][j] + bv[p * C_ + c]);
    }
}

// ---- kernel 7: pa = v·attn^T, write blended out -----------------------
// grid(16, 3, 96), block 128. out = part*(1+wp) + wp*g_pa*pa
__global__ void pa_kernel(const float* __restrict__ x,
                          const _Float16* __restrict__ v,
                          const _Float16* __restrict__ attn,
                          const float* __restrict__ gamma_pa,
                          const float* __restrict__ gate,
                          float* __restrict__ out) {
  int mt = (int)blockIdx.x, nc = (int)blockIdx.y, inst = (int)blockIdx.z;
  int b = inst / P_, p = inst % P_;
  int wave = (int)(threadIdx.x >> 5);
  const _Float16* vp = v + (size_t)inst * C_ * N_;
  const _Float16* ap = attn + (size_t)inst * N_ * N_;
  int m0 = mt * 64 + wave * 16;   // c rows
  int n0 = nc * 128;              // spatial cols (attn rows)

  v8f acc[8];
#pragma unroll
  for (int t = 0; t < 8; ++t)
#pragma unroll
    for (int j = 0; j < 8; ++j) acc[t][j] = 0.f;

  for (int k0 = 0; k0 < N_; k0 += 32) {
    __builtin_prefetch(vp + (size_t)m0 * N_ + k0 + 128, 0, 0);
    v16h a = loadA_f16(vp, N_, m0, k0);
#pragma unroll
    for (int t = 0; t < 8; ++t) {
      v16h bb = loadB_f16(ap, N_, n0 + t * 16, k0);
      acc[t] = wmma(a, bb, acc[t]);
    }
  }
  int l = laneid(), colb = l & 15, rofs = (l < 16) ? 0 : 8;
  float wp = gate[inst], g = gamma_pa[p];
  const float* xpart = x + (size_t)b * C_ * HW_ + (size_t)p * PH_ * W_;
  float* opart = out + (size_t)b * C_ * HW_ + (size_t)p * PH_ * W_;
#pragma unroll
  for (int t = 0; t < 8; ++t)
#pragma unroll
    for (int j = 0; j < 8; ++j) {
      int c = m0 + j + rofs, n = n0 + t * 16 + colb;
      size_t idx = (size_t)c * HW_ + n;
      opart[idx] = xpart[idx] * (1.f + wp) + wp * g * acc[t][j];
    }
}

// ---- kernel 8: fused CAM per 16-row tile ------------------------------
// grid(64, 96), block 256 (8 waves). LDS ~57KB.
// TDM-stage 16 xf rows -> E in regs -> cross-wave min/sum -> softmax(-E)
// -> A f16 in LDS -> ca = A·xf -> out += wp*g_ca*ca
__global__ void cam_kernel(const float* __restrict__ x,
                           const _Float16* __restrict__ xT,
                           const float* __restrict__ gamma_ca,
                           const float* __restrict__ gate,
                           float* __restrict__ out) {
  __shared__ float Stage[16][N_];      // 24KB staging; reused as f16 rows
  __shared__ _Float16 Att[16][C_];     // 32KB softmax attention rows
  __shared__ float red[8][16];
  __shared__ float rowMin[16];
  __shared__ float rowSum[16];

  int ct = (int)blockIdx.x, inst = (int)blockIdx.y;
  int b = inst / P_, p = inst % P_;
  int tid = (int)threadIdx.x, wave = tid >> 5, l = tid & 31;
  int colb = l & 15, rofs = (l < 16) ? 0 : 8;
  const float* xpart = x + (size_t)b * C_ * HW_ + (size_t)p * PH_ * W_;
  const _Float16* xTp = xT + (size_t)inst * N_ * C_;
  int c0 = ct * 16;
  const float* gsrc = xpart + (size_t)c0 * HW_;

#if defined(HAVE_TDM)
  // Tensor Data Mover: 2D tile, 4-byte elems, 384 x 16 rows, row stride 1152.
  if (wave == 0) {
    unsigned long long ga = (unsigned long long)(uintptr_t)gsrc;
    unsigned lds = (unsigned)(uintptr_t)&Stage[0][0];
    u32x4 g0;
    g0[0] = 1u;                                          // count=1, user mode
    g0[1] = lds;                                         // lds_addr
    g0[2] = (unsigned)(ga & 0xffffffffu);                // global_addr lo
    g0[3] = (unsigned)((ga >> 32) & 0x01ffffffu) | (2u << 30);  // hi + type=2
    i32x8 g1;
    g1[0] = (int)(2u << 16);          // data_size = 4B
    g1[1] = 0;                        // tensor_dim0 = 1<<20 (low16 here = 0)
    g1[2] = 16;                       // tensor_dim0 hi16 = 0x10, tensor_dim1 lo = 0
    g1[3] = (int)((384u << 16) | 1u); // tile_dim0 = 384, tensor_dim1 hi = 1
    g1[4] = 16;                       // tile_dim1 = 16
    g1[5] = HW_;                      // tensor_dim0_stride = 1152
    g1[6] = 0;
    g1[7] = 0;
    i32x4 gz4 = {0, 0, 0, 0};
    i32x8 gz8 = {0, 0, 0, 0, 0, 0, 0, 0};
    __builtin_amdgcn_tensor_load_to_lds(g0, g1, gz4, gz4, gz8, 0);
    __builtin_amdgcn_s_wait_tensorcnt(0);
  }
  __syncthreads();
#else
  for (int i = tid; i < 16 * N_; i += 256) {
    int r = i / N_, n = i % N_;
    Stage[r][n] = gsrc[(size_t)r * HW_ + n];
  }
  __syncthreads();
#endif

  // convert staged f32 rows to f16 in place (read all -> sync -> write)
  float tmp[24];
#pragma unroll
  for (int i = 0; i < 24; ++i) tmp[i] = ((const float*)&Stage[0][0])[tid + 256 * i];
  __syncthreads();
  _Float16* Ar = (_Float16*)&Stage[0][0];
#pragma unroll
  for (int i = 0; i < 24; ++i) Ar[tid + 256 * i] = (_Float16)tmp[i];
  __syncthreads();

  // phase 1: E[16][wave's 128 d-cols] in registers
  int d0 = wave * 128;
  v8f acc[8];
#pragma unroll
  for (int t = 0; t < 8; ++t)
#pragma unroll
    for (int j = 0; j < 8; ++j) acc[t][j] = 0.f;

  for (int k0 = 0; k0 < N_; k0 += 32) {
    v16h a = loadA_f16(Ar, N_, 0, k0);
#pragma unroll
    for (int t = 0; t < 8; ++t) {
      v16h bb = loadB_f32(xpart, HW_, d0 + t * 16, k0);  // B^T[d][n] = xf
      acc[t] = wmma(a, bb, acc[t]);
    }
  }

  // phase 2a: row min (softmax(-E) shift) across wave, then across waves
  float mn[8];
#pragma unroll
  for (int j = 0; j < 8; ++j) {
    mn[j] = acc[0][j];
#pragma unroll
    for (int t = 1; t < 8; ++t) mn[j] = fminf(mn[j], acc[t][j]);
  }
#pragma unroll
  for (int d = 8; d; d >>= 1)
#pragma unroll
    for (int j = 0; j < 8; ++j) mn[j] = fminf(mn[j], __shfl_xor(mn[j], d, 32));
  if ((l & 15) == 0)
#pragma unroll
    for (int j = 0; j < 8; ++j) red[wave][j + rofs] = mn[j];
  __syncthreads();
  if (tid < 16) {
    float v = red[0][tid];
#pragma unroll
    for (int w = 1; w < 8; ++w) v = fminf(v, red[w][tid]);
    rowMin[tid] = v;
  }
  __syncthreads();

  // phase 2b: exp(min - E), partial sums, cross-wave sum
  float sm[8];
#pragma unroll
  for (int j = 0; j < 8; ++j) {
    float rm = rowMin[j + rofs];
    float s = 0.f;
#pragma unroll
    for (int t = 0; t < 8; ++t) {
      float e = __expf(rm - acc[t][j]);
      acc[t][j] = e;
      s += e;
    }
    sm[j] = s;
  }
#pragma unroll
  for (int d = 8; d; d >>= 1)
#pragma unroll
    for (int j = 0; j < 8; ++j) sm[j] += __shfl_xor(sm[j], d, 32);
  if ((l & 15) == 0)
#pragma unroll
    for (int j = 0; j < 8; ++j) red[wave][j + rofs] = sm[j];
  __syncthreads();
  if (tid < 16) {
    float v = 0.f;
#pragma unroll
    for (int w = 0; w < 8; ++w) v += red[w][tid];
    rowSum[tid] = v;
  }
  __syncthreads();

  // phase 2c: normalize and stash A as f16
#pragma unroll
  for (int j = 0; j < 8; ++j) {
    int row = j + rofs;
    float inv = 1.f / rowSum[row];
#pragma unroll
    for (int t = 0; t < 8; ++t)
      Att[row][d0 + t * 16 + colb] = (_Float16)(acc[t][j] * inv);
  }
  __syncthreads();

  // phase 3: ca[16][48 cols per wave] = A · xf  (B from staged xf^T f16)
  v8f acc3[3];
#pragma unroll
  for (int t = 0; t < 3; ++t)
#pragma unroll
    for (int j = 0; j < 8; ++j) acc3[t][j] = 0.f;
  int n0 = wave * 48;
  for (int k0 = 0; k0 < C_; k0 += 32) {
    v16h a = loadA_f16(&Att[0][0], C_, 0, k0);
#pragma unroll
    for (int t = 0; t < 3; ++t) {
      v16h bb = loadB_f16(xTp, C_, n0 + t * 16, k0);
      acc3[t] = wmma(a, bb, acc3[t]);
    }
  }

  // phase 4: out += wp * g_ca * ca
  float wp = gate[inst], g = gamma_ca[p];
  float* opart = out + (size_t)b * C_ * HW_ + (size_t)p * PH_ * W_;
#pragma unroll
  for (int t = 0; t < 3; ++t)
#pragma unroll
    for (int j = 0; j < 8; ++j) {
      int c = c0 + j + rofs, n = n0 + t * 16 + colb;
      size_t idx = (size_t)c * HW_ + n;
      opart[idx] += wp * g * acc3[t][j];
    }
}

// ---- host launch -------------------------------------------------------
extern "C" void kernel_launch(void* const* d_in, const int* in_sizes, int n_in,
                              void* d_out, int out_size, void* d_ws, size_t ws_size,
                              hipStream_t stream) {
  const float* x        = (const float*)d_in[0];
  const int*   modality = (const int*)d_in[1];
  const float* Wq  = (const float*)d_in[2];
  const float* bq  = (const float*)d_in[3];
  const float* Wk  = (const float*)d_in[4];
  const float* bk  = (const float*)d_in[5];
  const float* Wv  = (const float*)d_in[6];
  const float* bv  = (const float*)d_in[7];
  const float* gpa = (const float*)d_in[8];
  const float* gca = (const float*)d_in[9];
  const float* gW1 = (const float*)d_in[10];
  const float* gb1 = (const float*)d_in[11];
  const float* gW2 = (const float*)d_in[12];
  const float* gb2 = (const float*)d_in[13];
  float* out = (float*)d_out;

  // ---- workspace layout (bytes), stream-serialized reuse ----
  char* ws = (char*)d_ws;
  size_t off = 0;
  float*    gate  = (float*)(ws + off);            off += 512;
  _Float16* xT    = (_Float16*)(ws + off);         off += (size_t)NI_ * N_ * C_ * 2;  // 75.5MB
  _Float16* wq16  = (_Float16*)(ws + off);         off += (size_t)P_ * IC_ * C_ * 2;
  _Float16* wk16  = (_Float16*)(ws + off);         off += (size_t)P_ * IC_ * C_ * 2;
  _Float16* wv16  = (_Float16*)(ws + off);         off += (size_t)P_ * C_ * C_ * 2;
  _Float16* bufA  = (_Float16*)(ws + off);         off += (size_t)NI_ * C_ * N_ * 2;  // 75.5MB
  _Float16* attn  = (_Float16*)(ws + off);         off += (size_t)NI_ * N_ * N_ * 2;  // 28.3MB
  _Float16* q     = bufA;                          // [96][384][128] f16
  _Float16* kT    = bufA + (size_t)NI_ * N_ * IC_; // [96][384][128] f16
  _Float16* v     = bufA;                          // [96][1024][384] f16 (reuse after attn)

  gate_kernel<<<1, 128, 0, stream>>>(modality, gW1, gb1, gW2, gb2, gate);
  cvt_f32_f16_kernel<<<192, 256, 0, stream>>>(Wq, wq16, P_ * IC_ * C_);
  cvt_f32_f16_kernel<<<192, 256, 0, stream>>>(Wk, wk16, P_ * IC_ * C_);
  cvt_f32_f16_kernel<<<1536, 256, 0, stream>>>(Wv, wv16, P_ * C_ * C_);
  stage_xT_kernel<<<dim3(16, 6, NI_), 256, 0, stream>>>(x, xT);

  proj_qk_kernel<<<dim3(6, NI_), 128, 0, stream>>>(xT, wq16, bq, q);
  proj_qk_kernel<<<dim3(6, NI_), 128, 0, stream>>>(xT, wk16, bk, kT);
  attn_kernel<<<dim3(12, NI_), 64, 0, stream>>>(q, kT, attn);
  proj_v_kernel<<<dim3(16, 3, NI_), 128, 0, stream>>>(xT, wv16, bv, v);
  pa_kernel<<<dim3(16, 3, NI_), 128, 0, stream>>>(x, v, attn, gpa, gate, out);
  cam_kernel<<<dim3(64, NI_), 256, 0, stream>>>(x, xT, gca, gate, out);
}
